// MultiHeadAttention_42528766165561
// MI455X (gfx1250) — matrix operand order repaired and checked by reading, not compiled, not run on hardware
//
#include <hip/hip_runtime.h>
#include <hip/hip_bf16.h>

// MI455X / gfx1250: wave32, WMMA 16x16x32 f16 -> f32.
// Pipeline: (1) QKV projection GEMMs -> f16 workspace (V stored transposed),
// (2) fused flash attention, 32 queries/block, scores never leave LDS,
// (3) output projection. GEMMs use 128x64 block tiles with register
// double-buffering of the K-panel staging. All WMMA fragments are contiguous
// 32B vector loads, hoisted ahead of the WMMA stream for latency cover.

typedef __attribute__((ext_vector_type(16))) _Float16 v16h;
typedef __attribute__((ext_vector_type(8)))  float    v8f;

#define B_  2
#define S_  2048
#define D_  1024
#define H_  16
#define DH_ 64
#define M_  (B_ * S_)      // 4096 rows

// ---------------------------------------------------------------------------
// WMMA helpers (layouts per CDNA5 ISA 7.12.2, wave32)
// ---------------------------------------------------------------------------

static __device__ __forceinline__ v8f wmma_f16(v16h a, v16h b, v8f c) {
  return __builtin_amdgcn_wmma_f32_16x16x32_f16(false, a, false, b, (short)0,
                                                c, false, false);
}

// A-matrix 16x32 f16 fragment. Lane holds row (lane&15). For lane-half `hi`:
// halves 0..15 <- K = hi*16 + j.
static __device__ __forceinline__ v16h load_a_pair(const _Float16* rowp, int hi) {
  v16h a;
  const _Float16* p0 = rowp + hi * 16;
#pragma unroll
  for (int j = 0; j < 16; ++j) a[j] = p0[j];
  return a;
}

static __device__ __forceinline__ v8f v8f_zero() {
  return (v8f){0.f, 0.f, 0.f, 0.f, 0.f, 0.f, 0.f, 0.f};
}

// ---------------------------------------------------------------------------
// Kernel 1: QKV projection.  GEMM M=4096, N=1024 (h*64+e), K=1024.
// 128x64 block tile, 4 waves, each wave owns 32 rows x 64 cols (8 WMMA accs).
// Register double-buffering of K-panel staging.
// Q,K outputs: [B,H,S,DH].  V output: [B,H,DH,S] (transposed for attention).
// ---------------------------------------------------------------------------
__global__ __launch_bounds__(128) void proj_qkv_kernel(
    const float* __restrict__ Q, const float* __restrict__ K,
    const float* __restrict__ V,
    const float* __restrict__ Wq, const float* __restrict__ bq,
    const float* __restrict__ Wk, const float* __restrict__ bk,
    const float* __restrict__ Wv, const float* __restrict__ bv,
    _Float16* __restrict__ Qh, _Float16* __restrict__ Kh,
    _Float16* __restrict__ VhT)
{
  const int tid  = threadIdx.x;
  const int lane = tid & 31;
  const int wave = tid >> 5;
  const int hi   = (lane >> 4) & 1;
  const int ln   = lane & 15;

  const int m0 = blockIdx.x * 128;
  const int n0 = blockIdx.y * 64;      // 64-wide tile == exactly one head
  const int h  = n0 >> 6;
  const int which = blockIdx.z;        // 0=Q 1=K 2=V

  const float* X    = (which == 0) ? Q  : (which == 1) ? K  : V;
  const float* W    = (which == 0) ? Wq : (which == 1) ? Wk : Wv;
  const float* bias = (which == 0) ? bq : (which == 1) ? bk : bv;
  _Float16*    out  = (which == 0) ? Qh : (which == 1) ? Kh : VhT;

  __shared__ _Float16 At [128][48];    // 128x32 A tile (96B rows)
  __shared__ _Float16 BtT[64][48];     // 32x64 B tile stored TRANSPOSED [n][k]

  v8f acc[2][4];
#pragma unroll
  for (int qt = 0; qt < 2; ++qt)
#pragma unroll
    for (int i = 0; i < 4; ++i) acc[qt][i] = v8f_zero();

  // Staging mapping: A -> thread owns full 32-wide row `tid`;
  // B -> thread owns column bn, k-half bk0 (coalesced across threads).
  const int bn  = tid & 63;
  const int bk0 = (tid >> 6) * 16;
  const float* Xrow = X + (size_t)(m0 + tid) * D_;
  const float* Wcol = W + (size_t)h * D_ * DH_ + bn;

  v16h pa0, pa1, pb;
  auto stage_load = [&](int k0) {
    const float* s = Xrow + k0;
#pragma unroll
    for (int j = 0; j < 16; ++j) { pa0[j] = (_Float16)s[j];
                                   pa1[j] = (_Float16)s[16 + j]; }
    const float* w = Wcol + (size_t)(k0 + bk0) * DH_;
#pragma unroll
    for (int j = 0; j < 16; ++j) pb[j] = (_Float16)w[(size_t)j * DH_];
  };

  stage_load(0);
  for (int k0 = 0; k0 < D_; k0 += 32) {
    *(v16h*)&At[tid][0]  = pa0;
    *(v16h*)&At[tid][16] = pa1;
    *(v16h*)&BtT[bn][bk0] = pb;
    __syncthreads();
    if (k0 + 32 < D_) stage_load(k0 + 32);   // overlap next panel w/ compute

    // Hoist all fragments: 12 ds_load_b128 batched ahead of the WMMA stream.
    const v16h a0 = load_a_pair(&At[wave * 32 + ln][0], hi);
    const v16h a1 = load_a_pair(&At[wave * 32 + 16 + ln][0], hi);
    v16h bfr[4];
#pragma unroll
    for (int nt = 0; nt < 4; ++nt)
      bfr[nt] = *(const v16h*)&BtT[nt * 16 + ln][hi * 16];
#pragma unroll
    for (int nt = 0; nt < 4; ++nt) {
      acc[0][nt] = wmma_f16(a0, bfr[nt], acc[0][nt]);
      acc[1][nt] = wmma_f16(a1, bfr[nt], acc[1][nt]);
    }
    __syncthreads();
  }

  // Epilogue: bias add, store f16. Q/K -> [B,H,S,DH]; V -> [B,H,DH,S].
#pragma unroll
  for (int qt = 0; qt < 2; ++qt) {
#pragma unroll
    for (int nt = 0; nt < 4; ++nt) {
      const int n  = n0 + nt * 16 + ln;
      const float bval = bias[n];
      const int e  = n & (DH_ - 1);
#pragma unroll
      for (int r = 0; r < 8; ++r) {
        const int m  = m0 + wave * 32 + qt * 16 + r + hi * 8;
        const int bb = m >> 11;          // m / S_
        const int s  = m & (S_ - 1);
        const float val = acc[qt][nt][r] + bval;
        if (which == 2)
          out[(((size_t)bb * H_ + h) * DH_ + e) * S_ + s] = (_Float16)val;
        else
          out[(((size_t)bb * H_ + h) * S_ + s) * DH_ + e] = (_Float16)val;
      }
    }
  }
}

// ---------------------------------------------------------------------------
// Kernel 2: fused attention.  One block = (b, h, 32 queries).
// Key chunks of 256, online softmax, scores live only in LDS.
// K^T and V fragments load straight from global as contiguous 32B vectors.
// ---------------------------------------------------------------------------
#define TQ 32
#define TC 256

__global__ __launch_bounds__(128) void attn_kernel(
    const _Float16* __restrict__ Qh, const _Float16* __restrict__ Kh,
    const _Float16* __restrict__ VhT, _Float16* __restrict__ Out)
{
  const int tid  = threadIdx.x;
  const int lane = tid & 31;
  const int wave = tid >> 5;
  const int hi   = (lane >> 4) & 1;
  const int ln   = lane & 15;

  const int q0 = blockIdx.x * TQ;
  const int h  = blockIdx.y;
  const int bb = blockIdx.z;

  __shared__ float    sc[TQ][264];     // score chunk 32x256 (padded)
  __shared__ _Float16 pf[TQ][264];     // probabilities f16 (rows 16B aligned)
  __shared__ float redmax[TQ][4];
  __shared__ float redsum[TQ][4];
  __shared__ float m_run[TQ], l_run[TQ], alpha[TQ];

  if (tid < TQ) { m_run[tid] = -1e30f; l_run[tid] = 0.f; }
  __syncthreads();

  const size_t headbase = ((size_t)bb * H_ + h) * S_ * DH_;  // == *DH_*S_

  // Q A-fragments for both 16-row tiles, reused for every key tile.
  v16h qa[2][2];
#pragma unroll
  for (int qt = 0; qt < 2; ++qt) {
    const _Float16* qrow = Qh + headbase + (size_t)(q0 + qt * 16 + ln) * DH_;
    qa[qt][0] = load_a_pair(qrow, hi);
    qa[qt][1] = load_a_pair(qrow + 32, hi);
  }

  // V column for this lane: dh = wave*16 + ln, contiguous along keys.
  const _Float16* vcol = VhT + headbase + (size_t)(wave * 16 + ln) * S_;

  v8f acc[2] = {v8f_zero(), v8f_zero()};
  const float scale = 0.125f;          // 1/sqrt(DK=64)

  for (int c = 0; c < S_ / TC; ++c) {
    const int t0 = c * TC;

    // ---- scores: each wave computes 4 key tiles x 2 query tiles ----
#pragma unroll
    for (int i = 0; i < 4; ++i) {
      const int kt  = wave * 4 + i;
      const int key = t0 + kt * 16 + ln;
      const _Float16* kcol = Kh + headbase + (size_t)key * DH_;
      const v16h b0 = *(const v16h*)(kcol + hi * 16);
      const v16h b1 = *(const v16h*)(kcol + 32 + hi * 16);
      if (c + 1 < S_ / TC)
        __builtin_prefetch(kcol + (size_t)TC * DH_, 0, 1);
#pragma unroll
      for (int qt = 0; qt < 2; ++qt) {
        v8f s = v8f_zero();
        s = wmma_f16(qa[qt][0], b0, s);
        s = wmma_f16(qa[qt][1], b1, s);
#pragma unroll
        for (int r = 0; r < 8; ++r)
          sc[qt * 16 + r + hi * 8][kt * 16 + ln] = s[r] * scale;
      }
    }
    __syncthreads();

    // ---- online softmax: 32 rows x 4 segments of 64 ----
    {
      const int row = tid & 31, seg = tid >> 5;
      float mx = -1e30f;
      for (int j = 0; j < 64; ++j) mx = fmaxf(mx, sc[row][seg * 64 + j]);
      redmax[row][seg] = mx;
    }
    __syncthreads();
    if (tid < TQ) {
      float cm = fmaxf(fmaxf(redmax[tid][0], redmax[tid][1]),
                       fmaxf(redmax[tid][2], redmax[tid][3]));
      const float nm = fmaxf(m_run[tid], cm);
      alpha[tid] = __expf(m_run[tid] - nm);
      m_run[tid] = nm;
    }
    __syncthreads();
    {
      const int row = tid & 31, seg = tid >> 5;
      const float mr = m_run[row];
      float sm = 0.f;
      for (int j = 0; j < 64; ++j) {
        const float p = __expf(sc[row][seg * 64 + j] - mr);
        pf[row][seg * 64 + j] = (_Float16)p;
        sm += p;
      }
      redsum[row][seg] = sm;
    }
    __syncthreads();
    if (tid < TQ) {
      const float sm = (redsum[tid][0] + redsum[tid][1]) +
                       (redsum[tid][2] + redsum[tid][3]);
      l_run[tid] = l_run[tid] * alpha[tid] + sm;
    }
    __syncthreads();

    // ---- P @ V: wave owns dh columns [wave*16, wave*16+16) ----
#pragma unroll
    for (int qt = 0; qt < 2; ++qt)
#pragma unroll
      for (int r = 0; r < 8; ++r) acc[qt][r] *= alpha[qt * 16 + r + hi * 8];
#pragma unroll
    for (int ks = 0; ks < 8; ++ks) {
      // Hoist all three operands before the WMMA pair.
      const v16h b  = *(const v16h*)(vcol + t0 + ks * 32 + hi * 16);
      const v16h a0 = load_a_pair(&pf[ln][ks * 32], hi);
      const v16h a1 = load_a_pair(&pf[16 + ln][ks * 32], hi);
      acc[0] = wmma_f16(a0, b, acc[0]);
      acc[1] = wmma_f16(a1, b, acc[1]);
    }
    __syncthreads();   // protect sc/pf before next chunk overwrites
  }

  // ---- epilogue: normalize, store concat layout [B,S,H*DH] f16 ----
#pragma unroll
  for (int qt = 0; qt < 2; ++qt) {
#pragma unroll
    for (int r = 0; r < 8; ++r) {
      const int mrow = qt * 16 + r + hi * 8;
      const float inv = 1.f / l_run[mrow];
      const size_t m = (size_t)bb * S_ + q0 + mrow;
      Out[m * D_ + h * DH_ + wave * 16 + ln] = (_Float16)(acc[qt][r] * inv);
    }
  }
}

// ---------------------------------------------------------------------------
// Kernel 3: output projection.  out[m,n] = concat[m,:] . Wf[:,n] + bf[n], f32
// 128x64 block tile, register double-buffered staging.
// ---------------------------------------------------------------------------
__global__ __launch_bounds__(128) void final_proj_kernel(
    const _Float16* __restrict__ A, const float* __restrict__ Wf,
    const float* __restrict__ bf, float* __restrict__ Out)
{
  const int tid  = threadIdx.x;
  const int lane = tid & 31;
  const int wave = tid >> 5;
  const int hi   = (lane >> 4) & 1;
  const int ln   = lane & 15;

  const int m0 = blockIdx.x * 128;
  const int n0 = blockIdx.y * 64;

  __shared__ _Float16 At [128][48];
  __shared__ _Float16 BtT[64][48];     // B tile transposed [n][k]

  v8f acc[2][4];
#pragma unroll
  for (int qt = 0; qt < 2; ++qt)
#pragma unroll
    for (int i = 0; i < 4; ++i) acc[qt][i] = v8f_zero();

  const int bn  = tid & 63;
  const int bk0 = (tid >> 6) * 16;
  const _Float16* Arow = A + (size_t)(m0 + tid) * D_;
  const float*    Wcol = Wf + n0 + bn;

  v16h pa0, pa1, pb;
  auto stage_load = [&](int k0) {
    pa0 = *(const v16h*)(Arow + k0);
    pa1 = *(const v16h*)(Arow + k0 + 16);
    const float* w = Wcol + (size_t)(k0 + bk0) * D_;
#pragma unroll
    for (int j = 0; j < 16; ++j) pb[j] = (_Float16)w[(size_t)j * D_];
  };

  stage_load(0);
  for (int k0 = 0; k0 < D_; k0 += 32) {
    *(v16h*)&At[tid][0]  = pa0;
    *(v16h*)&At[tid][16] = pa1;
    *(v16h*)&BtT[bn][bk0] = pb;
    __syncthreads();
    if (k0 + 32 < D_) stage_load(k0 + 32);

    const v16h a0 = load_a_pair(&At[wave * 32 + ln][0], hi);
    const v16h a1 = load_a_pair(&At[wave * 32 + 16 + ln][0], hi);
    v16h bfr[4];
#pragma unroll
    for (int nt = 0; nt < 4; ++nt)
      bfr[nt] = *(const v16h*)&BtT[nt * 16 + ln][hi * 16];
#pragma unroll
    for (int nt = 0; nt < 4; ++nt) {
      acc[0][nt] = wmma_f16(a0, bfr[nt], acc[0][nt]);
      acc[1][nt] = wmma_f16(a1, bfr[nt], acc[1][nt]);
    }
    __syncthreads();
  }

#pragma unroll
  for (int qt = 0; qt < 2; ++qt) {
#pragma unroll
    for (int nt = 0; nt < 4; ++nt) {
      const int n = n0 + nt * 16 + ln;
      const float bval = bf[n];
#pragma unroll
      for (int r = 0; r < 8; ++r) {
        const int m = m0 + wave * 32 + qt * 16 + r + hi * 8;
        Out[(size_t)m * D_ + n] = acc[qt][nt][r] + bval;
      }
    }
  }
}

// ---------------------------------------------------------------------------
// Host launcher
// ---------------------------------------------------------------------------
extern "C" void kernel_launch(void* const* d_in, const int* in_sizes, int n_in,
                              void* d_out, int out_size, void* d_ws, size_t ws_size,
                              hipStream_t stream) {
  (void)in_sizes; (void)n_in; (void)out_size;

  const float* Q  = (const float*)d_in[0];
  const float* K  = (const float*)d_in[1];
  const float* V  = (const float*)d_in[2];
  const float* Wq = (const float*)d_in[3];
  const float* bq = (const float*)d_in[4];
  const float* Wk = (const float*)d_in[5];
  const float* bk = (const float*)d_in[6];
  const float* Wv = (const float*)d_in[7];
  const float* bv = (const float*)d_in[8];
  const float* Wf = (const float*)d_in[9];
  const float* bf = (const float*)d_in[10];
  float* out = (float*)d_out;

  // Workspace: Qh, Kh, VhT, concat — each B*H*S*DH f16 = 8 MB; 32 MB total.
  const size_t headElems = (size_t)B_ * H_ * S_ * DH_;   // 4M halves
  _Float16* qh  = (_Float16*)d_ws;
  _Float16* kh  = qh + headElems;
  _Float16* vht = kh + headElems;
  _Float16* ao  = vht + headElems;
  (void)ws_size;  // requires >= 32 MB

  dim3 blk(128);
  proj_qkv_kernel<<<dim3(M_ / 128, D_ / 64, 3), blk, 0, stream>>>(
      Q, K, V, Wq, bq, Wk, bk, Wv, bv, qh, kh, vht);
  attn_kernel<<<dim3(S_ / TQ, H_, B_), blk, 0, stream>>>(qh, kh, vht, ao);
  final_proj_kernel<<<dim3(M_ / 128, D_ / 64), blk, 0, stream>>>(ao, Wf, bf, out);
}
